// DotProductCausalRotaryAttention_38620345925719
// MI455X (gfx1250) — matrix operand order repaired and checked by reading, not compiled, not run on hardware
//
#include <hip/hip_runtime.h>

typedef __attribute__((ext_vector_type(16))) _Float16 v16h;
typedef __attribute__((ext_vector_type(8)))  float    v8f;
typedef __attribute__((ext_vector_type(4)))  _Float16 v4h;
typedef __attribute__((ext_vector_type(2)))  _Float16 v2h;

#define SEQ   2048
#define DH    128
#define NBH   32      // B*H
#define ROT   32      // rotary dim
#define BQ    128     // q rows per block (8 waves x 16)
#define BK    64      // key tile
#define PADD  8

union ABReg { v16h h; uint4 u[2]; };

__global__ __launch_bounds__(256, 1)
void fattn_rotary_wmma(const float* __restrict__ Qg,
                       const float* __restrict__ Kg,
                       const float* __restrict__ Vg,
                       float* __restrict__ Og)
{
    // k_sh/v_sh are aliased by the (phase-0 only) Q^T staging buffer:
    //   qT view : [DH][BQ+PADD]  = 128*136 halfs = 34816 B
    //   k_sh    : [BK][DH+PADD]  =  64*136 halfs
    //   v_sh    : [BK][DH+PADD]  =  64*136 halfs   (sum = 34816 B, exact fit)
    __shared__ _Float16 smem[2 * BK * (DH + PADD)];
    __shared__ _Float16 p_sh[8][16][BK + PADD];    // per-wave P^T tile [q][key]

    _Float16 (*qT_sh)[BQ + PADD] = reinterpret_cast<_Float16(*)[BQ + PADD]>(smem);
    _Float16 (*k_sh)[DH + PADD]  = reinterpret_cast<_Float16(*)[DH + PADD]>(smem);
    _Float16 (*v_sh)[DH + PADD]  = reinterpret_cast<_Float16(*)[DH + PADD]>(smem + BK * (DH + PADD));

    const int tid  = threadIdx.x;
    const int wave = tid >> 5;
    const int lane = tid & 31;
    const int lrow = lane & 15;
    const int hi   = lane >> 4;

    const int qtile = blockIdx.x;
    const int bh    = blockIdx.y;
    const int q0    = qtile * BQ;

    const size_t base = (size_t)bh * SEQ * DH;
    const float scale = 0.08838834764831845f;   // 1/sqrt(128)
    const float nlf   = 0.5756462732485114f;    // ln(10000)/16

    // ---------- phase 0: Q tile -> qT_sh (transposed, rotary + scale) ----------
    #pragma unroll
    for (int j = 0; j < 8; ++j) {               // 128 rows * 16 pairs / 256
        int p   = tid + j * 256;
        int row = p >> 4;
        int i   = p & 15;
        float invf = __expf(-nlf * (float)i);
        float ang  = (float)(q0 + row) * invf;
        float sn, cs; __sincosf(ang, &sn, &cs);
        const float* qp = Qg + base + (size_t)(q0 + row) * DH + 2 * i;
        float x0 = qp[0], x1 = qp[1];
        qT_sh[2*i  ][row] = (_Float16)((x0 * cs - x1 * sn) * scale);
        qT_sh[2*i+1][row] = (_Float16)((x1 * cs + x0 * sn) * scale);
    }
    #pragma unroll
    for (int j = 0; j < 12; ++j) {              // 128 rows * 96 dims / 4 / 256
        int p   = tid + j * 256;
        int row = p / 24;
        int d   = ROT + 4 * (p % 24);
        const float4 x = *(const float4*)(Qg + base + (size_t)(q0 + row) * DH + d);
        qT_sh[d  ][row] = (_Float16)(x.x * scale);
        qT_sh[d+1][row] = (_Float16)(x.y * scale);
        qT_sh[d+2][row] = (_Float16)(x.z * scale);
        qT_sh[d+3][row] = (_Float16)(x.w * scale);
    }
    __syncthreads();

    // ---------- persistent Q^T B-fragments (lane = q col, 16 dims in table k order) ----------
    ABReg bQ[4];
    #pragma unroll
    for (int c = 0; c < 4; ++c) {
        #pragma unroll
        for (int s = 0; s < 16; ++s)
            bQ[c].h[s] = qT_sh[32*c + (s < 8 ? 8*hi + s : 8 + 8*hi + s)][wave * 16 + lrow];
    }
    __syncthreads();   // qT reads done before k_sh/v_sh overwrite the buffer

    v8f acc[8];
    #pragma unroll
    for (int b = 0; b < 8; ++b) acc[b] = (v8f){0,0,0,0,0,0,0,0};
    float m = -3.0e38f, l = 0.0f;               // stats for q-col = lrow

    const int ktmax = 2 * qtile + 1;            // keys up to q0 + BQ - 1
    for (int kt = 0; kt <= ktmax; ++kt) {
        __syncthreads();   // previous tile's LDS reads complete

        if (kt < ktmax) {  // hint-prefetch next K/V tile (covers full 64x512B tiles)
            const float* kp = Kg + base + (size_t)((kt+1)*BK + (tid >> 2)) * DH + (tid & 3) * 32;
            const float* vp = Vg + base + (size_t)((kt+1)*BK + (tid >> 2)) * DH + (tid & 3) * 32;
            __builtin_prefetch(kp, 0, 3);
            __builtin_prefetch(vp, 0, 3);
        }

        // ---- K tile -> k_sh (row-major, rotary on dims 0..31, packed stores) ----
        #pragma unroll
        for (int j = 0; j < 4; ++j) {           // 64 keys * 16 pairs / 256
            int p   = tid + j * 256;
            int key = p >> 4;
            int i   = p & 15;
            float invf = __expf(-nlf * (float)i);
            float ang  = (float)(kt * BK + key) * invf;
            float sn, cs; __sincosf(ang, &sn, &cs);
            const float* kp = Kg + base + (size_t)(kt*BK + key) * DH + 2*i;
            float x0 = kp[0], x1 = kp[1];
            v2h h;
            h.x = (_Float16)(x0 * cs - x1 * sn);
            h.y = (_Float16)(x1 * cs + x0 * sn);
            *(v2h*)&k_sh[key][2*i] = h;
        }
        #pragma unroll
        for (int j = 0; j < 6; ++j) {           // 64 keys * 96 dims / 4 / 256
            int p   = tid + j * 256;
            int key = p / 24;
            int d   = ROT + 4 * (p % 24);
            const float4 x = *(const float4*)(Kg + base + (size_t)(kt*BK + key) * DH + d);
            v4h h;
            h.x = (_Float16)x.x; h.y = (_Float16)x.y;
            h.z = (_Float16)x.z; h.w = (_Float16)x.w;
            *(v4h*)&k_sh[key][d] = h;
        }
        // ---- V tile -> v_sh (row-major) ----
        #pragma unroll
        for (int j = 0; j < 8; ++j) {           // 64 keys * 128 dims / 4 / 256
            int p   = tid + j * 256;
            int key = p >> 5;
            int d   = (p & 31) * 4;
            const float4 x = *(const float4*)(Vg + base + (size_t)(kt*BK + key) * DH + d);
            v4h h;
            h.x = (_Float16)x.x; h.y = (_Float16)x.y;
            h.z = (_Float16)x.z; h.w = (_Float16)x.w;
            *(v4h*)&v_sh[key][d] = h;
        }
        __syncthreads();

        // ---- S^T = K Q^T : 4 key-blocks x 4 k-chunks = 16 WMMA ----
        // C layout: VGPR r -> key = kb*16 + r + 8*hi ; N col -> q = lrow
        v8f sblk[4];
        #pragma unroll
        for (int kb = 0; kb < 4; ++kb) {
            v8f s = (v8f){0,0,0,0,0,0,0,0};
            #pragma unroll
            for (int c = 0; c < 4; ++c) {
                ABReg a;  // A = K rows (natural layout gather)
                const _Float16* kr = &k_sh[kb * 16 + lrow][32*c + hi*8];
                a.u[0] = *(const uint4*)kr;
                a.u[1] = *(const uint4*)(kr + 16);
                s = __builtin_amdgcn_wmma_f32_16x16x32_f16(
                        false, a.h, false, bQ[c].h, (short)0, s, false, false);
            }
            sblk[kb] = s;
        }

        // ---- causal mask (any tile overlapping / past the wave's q rows) ----
        const int qrow = q0 + wave * 16 + lrow;
        if (kt * BK + BK - 1 > q0 + wave * 16) {
            #pragma unroll
            for (int kb = 0; kb < 4; ++kb) {
                #pragma unroll
                for (int r = 0; r < 8; ++r) {
                    int key = kt * BK + kb * 16 + r + 8 * hi;
                    if (key > qrow) sblk[kb][r] = -1.0e30f;
                }
            }
        }

        // ---- online softmax: per-lane reduce over 32 keys + one shfl_xor(16) ----
        float mn = m;
        #pragma unroll
        for (int kb = 0; kb < 4; ++kb)
            #pragma unroll
            for (int r = 0; r < 8; ++r) mn = fmaxf(mn, sblk[kb][r]);
        mn = fmaxf(mn, __shfl_xor(mn, 16, 32));

        float alpha = __expf(m - mn);
        m = mn;
        float rs = 0.0f;
        #pragma unroll
        for (int kb = 0; kb < 4; ++kb)
            #pragma unroll
            for (int r = 0; r < 8; ++r) {
                float pv = __expf(sblk[kb][r] - m);
                rs += pv;
                p_sh[wave][lrow][kb * 16 + r + 8 * hi] = (_Float16)pv;
            }
        rs += __shfl_xor(rs, 16, 32);
        l = l * alpha + rs;

        // broadcast alpha from lane holding q = r + 8*hi, rescale accumulators
        float alphaB[8];
        #pragma unroll
        for (int r = 0; r < 8; ++r) alphaB[r] = __shfl(alpha, r + 8 * hi, 32);
        #pragma unroll
        for (int b = 0; b < 8; ++b)
            #pragma unroll
            for (int r = 0; r < 8; ++r) acc[b][r] *= alphaB[r];

        // ---- O += P V : 2 k-chunks x 8 d-blocks = 16 WMMA (double-buffered B) ----
        ABReg aP[2];
        {
            const _Float16* pr = &p_sh[wave][lrow][0];
            #pragma unroll
            for (int kc = 0; kc < 2; ++kc) {
                aP[kc].u[0] = *(const uint4*)(pr + 32*kc +      hi*8);
                aP[kc].u[1] = *(const uint4*)(pr + 32*kc + 16 + hi*8);
            }
        }
        ABReg bv[2];
        {
            #pragma unroll
            for (int s = 0; s < 16; ++s)
                bv[0].h[s] = v_sh[(s < 8 ? 8*hi + s : 8 + 8*hi + s)][lrow];
        }
        #pragma unroll
        for (int idx = 0; idx < 16; ++idx) {
            int kc = idx >> 3;
            int nb = idx & 7;
            if (idx < 15) {
                int kcn = (idx + 1) >> 3;
                int nbn = (idx + 1) & 7;
                #pragma unroll
                for (int s = 0; s < 16; ++s)
                    bv[(idx + 1) & 1].h[s] = v_sh[32 * kcn + (s < 8 ? 8*hi + s : 8 + 8*hi + s)][nbn * 16 + lrow];
            }
            acc[nb] = __builtin_amdgcn_wmma_f32_16x16x32_f16(
                          false, aP[kc].h, false, bv[idx & 1].h, (short)0, acc[nb], false, false);
        }
    }

    // ---------- epilogue: O / l (broadcast l per accumulator row) ----------
    float lB[8];
    #pragma unroll
    for (int r = 0; r < 8; ++r) lB[r] = __shfl(l, r + 8 * hi, 32);
    #pragma unroll
    for (int b = 0; b < 8; ++b) {
        #pragma unroll
        for (int r = 0; r < 8; ++r) {
            int row = q0 + wave * 16 + r + 8 * hi;
            int col = b * 16 + lrow;
            Og[base + (size_t)row * DH + col] = acc[b][r] / lB[r];
        }
    }
}

extern "C" void kernel_launch(void* const* d_in, const int* in_sizes, int n_in,
                              void* d_out, int out_size, void* d_ws, size_t ws_size,
                              hipStream_t stream) {
    const float* q = (const float*)d_in[0];
    const float* k = (const float*)d_in[1];
    const float* v = (const float*)d_in[2];
    float* o = (float*)d_out;
    (void)in_sizes; (void)n_in; (void)out_size; (void)d_ws; (void)ws_size;
    dim3 grid(SEQ / BQ, NBH);
    fattn_rotary_wmma<<<grid, dim3(256), 0, stream>>>(q, k, v, o);
}
